// GINNet_635655160279
// MI455X (gfx1250) — matrix operand order repaired and checked by reading, not compiled, not run on hardware
//
#include <hip/hip_runtime.h>

#define NNODES 100000
#define NEDGES 1600000
#define DIM    128
#define OUTCH  2

typedef __attribute__((ext_vector_type(16))) _Float16 v16h;
typedef __attribute__((ext_vector_type(8)))  float    v8f;

// ---------------------------------------------------------------- utilities
__global__ __launch_bounds__(256) void k_fill0(float* __restrict__ p, long n) {
    long i = (long)blockIdx.x * 256 + threadIdx.x;
    if (i < n) p[i] = 0.0f;
}

__global__ __launch_bounds__(256) void k_deg(const int* __restrict__ dst,
                                             float* __restrict__ deg, int nE) {
    int e = blockIdx.x * 256 + threadIdx.x;
    if (e < nE) unsafeAtomicAdd(&deg[dst[e]], 1.0f);
}

__global__ __launch_bounds__(256) void k_invdeg(float* __restrict__ deg, int n) {
    int i = blockIdx.x * 256 + threadIdx.x;
    if (i < n) deg[i] = 1.0f / fmaxf(deg[i], 1.0f);
}

// one wave (32 lanes) per edge: lane handles 4 consecutive floats of the row
__global__ __launch_bounds__(256) void k_scatter(const float* __restrict__ h,
                                                 const int* __restrict__ src,
                                                 const int* __restrict__ dst,
                                                 float* __restrict__ agg, int nE) {
    int e = blockIdx.x * 8 + (threadIdx.x >> 5);
    if (e >= nE) return;
    int lane = threadIdx.x & 31;
    const float4 v = *(const float4*)(h + (long)src[e] * DIM + lane * 4);
    float* d = agg + (long)dst[e] * DIM + lane * 4;
    unsafeAtomicAdd(d + 0, v.x);
    unsafeAtomicAdd(d + 1, v.y);
    unsafeAtomicAdd(d + 2, v.z);
    unsafeAtomicAdd(d + 3, v.w);
}

// z = (1+eps[l])*h + agg * inv_deg[row]
__global__ __launch_bounds__(256) void k_combine(const float* __restrict__ h,
                                                 const float* __restrict__ agg,
                                                 const float* __restrict__ invdeg,
                                                 const float* __restrict__ eps, int layer,
                                                 float* __restrict__ z, long n4) {
    long i = (long)blockIdx.x * 256 + threadIdx.x;   // float4 index
    if (i >= n4) return;
    float id = invdeg[i >> 5];                       // 32 float4 per 128-wide row
    float e1 = 1.0f + eps[layer];
    float4 hv = ((const float4*)h)[i];
    float4 av = ((const float4*)agg)[i];
    float4 r;
    r.x = e1 * hv.x + av.x * id;
    r.y = e1 * hv.y + av.y * id;
    r.z = e1 * hv.z + av.z * id;
    r.w = e1 * hv.w + av.w * id;
    ((float4*)z)[i] = r;
}

// ---------------------------------------------------------------- WMMA GEMM
// Out[r][n] = lrelu( sum_k Zin[r][k] * W[k][n] + bias[n] ),  K = N = 128
// Block = 256 threads = 8 waves; each wave computes a 16x128 strip.
__global__ __launch_bounds__(256) void k_gemm_lrelu(const float* __restrict__ Zin,
                                                    const float* __restrict__ W,
                                                    const float* __restrict__ bias,
                                                    float* __restrict__ Out, int nRows) {
    // W transposed to [n][k] in f16, padded to 144 halfs (288B: 16B-aligned rows)
    __shared__ _Float16 Wt[128 * 144];

    const int tid = threadIdx.x;
    for (int i = tid; i < 128 * 128; i += 256) {
        int k = i >> 7, n = i & 127;           // W is row-major [k][n]
        Wt[n * 144 + k] = (_Float16)W[i];
    }
    __syncthreads();

    const int wave = tid >> 5;
    const int lane = tid & 31;
    const int half = lane >> 4;                 // K-half selector per ISA layout
    const int l16  = lane & 15;
    const int rowBase = blockIdx.x * 128 + wave * 16;

    // ---- A fragments (16-bit A 16x32 layout):
    // lanes 0-15: m=lane, VGPR0-3 hold K=0..7, VGPR4-7 hold K=16..23
    // lanes 16-31: same m, K offsets +8 (low VGPRs) / +8 (high VGPRs)
    v16h afr[4];
    {
        long r = rowBase + l16;
        if (r >= nRows) r = nRows - 1;          // clamp; stores are masked
        const float* zp = Zin + r * DIM;
#pragma unroll
        for (int t = 0; t < 4; ++t) {
            const float* p0 = zp + 32 * t + (half ? 8 : 0);        // j=0..7
            const float* p1 = zp + 32 * t + 16 + (half ? 8 : 0);   // j=8..15
            float4 a0 = *(const float4*)(p0);
            float4 a1 = *(const float4*)(p0 + 4);
            float4 a2 = *(const float4*)(p1);
            float4 a3 = *(const float4*)(p1 + 4);
            afr[t][0]  = (_Float16)a0.x; afr[t][1]  = (_Float16)a0.y;
            afr[t][2]  = (_Float16)a0.z; afr[t][3]  = (_Float16)a0.w;
            afr[t][4]  = (_Float16)a1.x; afr[t][5]  = (_Float16)a1.y;
            afr[t][6]  = (_Float16)a1.z; afr[t][7]  = (_Float16)a1.w;
            afr[t][8]  = (_Float16)a2.x; afr[t][9]  = (_Float16)a2.y;
            afr[t][10] = (_Float16)a2.z; afr[t][11] = (_Float16)a2.w;
            afr[t][12] = (_Float16)a3.x; afr[t][13] = (_Float16)a3.y;
            afr[t][14] = (_Float16)a3.z; afr[t][15] = (_Float16)a3.w;
        }
    }

    v8f acc[8];
#pragma unroll
    for (int c = 0; c < 8; ++c) acc[c] = (v8f){0.f,0.f,0.f,0.f,0.f,0.f,0.f,0.f};

    // ---- main loop: K-step outer, 8 independent column tiles inner (ILP)
    // B layout (16-bit 32x16): lane n=l16, same K order as A: elems 0-7 -> K=8*half+0..7, 8-15 -> K=16+8*half+0..7
#pragma unroll
    for (int t = 0; t < 4; ++t) {
#pragma unroll
        for (int c = 0; c < 8; ++c) {
            v16h bfr;
            const _Float16* wp = &Wt[(16 * c + l16) * 144 + 32 * t + (half ? 8 : 0)];
            __builtin_memcpy(&bfr, wp, 16);     // 2x ds_load_b128
            __builtin_memcpy((_Float16*)&bfr + 8, wp + 16, 16);
            acc[c] = __builtin_amdgcn_wmma_f32_16x16x32_f16(
                false, afr[t], false, bfr, (short)0, acc[c], false, false);
        }
    }

    // ---- epilogue: bias + leaky ReLU; C/D layout: VGPR r -> row r + 8*half
#pragma unroll
    for (int c = 0; c < 8; ++c) {
        float bv = bias[16 * c + l16];
#pragma unroll
        for (int r8 = 0; r8 < 8; ++r8) {
            long row = rowBase + r8 + (half ? 8 : 0);
            if (row < nRows) {
                float v = acc[c][r8] + bv;
                Out[row * DIM + 16 * c + l16] = (v > 0.f) ? v : 0.01f * v;
            }
        }
    }
}

// ---------------------------------------------------------------- 128 -> 2 head
__global__ __launch_bounds__(256) void k_out(const float* __restrict__ h,
                                             const float* __restrict__ Wout,
                                             const float* __restrict__ bout,
                                             float* __restrict__ out, int n) {
    int node = blockIdx.x * 8 + (threadIdx.x >> 5);
    if (node >= n) return;
    int lane = threadIdx.x & 31;
    const float4 hv = *(const float4*)(h + (long)node * DIM + lane * 4);
    float s0 = 0.f, s1 = 0.f;
    const float* a = (const float*)&hv;
#pragma unroll
    for (int j = 0; j < 4; ++j) {
        int k = lane * 4 + j;
        s0 += a[j] * Wout[2 * k + 0];
        s1 += a[j] * Wout[2 * k + 1];
    }
    for (int off = 16; off; off >>= 1) {
        s0 += __shfl_down(s0, off, 32);
        s1 += __shfl_down(s1, off, 32);
    }
    if (lane == 0) {
        out[2 * node + 0] = s0 + bout[0];
        out[2 * node + 1] = s1 + bout[1];
    }
}

// ---------------------------------------------------------------- launcher
extern "C" void kernel_launch(void* const* d_in, const int* in_sizes, int n_in,
                              void* d_out, int out_size, void* d_ws, size_t ws_size,
                              hipStream_t stream) {
    const float* x    = (const float*)d_in[0];
    const int*   src  = (const int*)d_in[1];
    const int*   dst  = (const int*)d_in[2];
    const float* W1   = (const float*)d_in[3];
    const float* b1   = (const float*)d_in[4];
    const float* W2   = (const float*)d_in[5];
    const float* b2   = (const float*)d_in[6];
    const float* eps  = (const float*)d_in[7];
    const float* Wout = (const float*)d_in[8];
    const float* bout = (const float*)d_in[9];
    float* out = (float*)d_out;

    const int  N = NNODES, E = NEDGES;
    const long ND = (long)N * DIM;

    float* bufA = (float*)d_ws;          // N*D
    float* agg  = bufA + ND;             // N*D
    float* deg  = agg + ND;              // N
    float* nemb = out + (long)N * OUTCH; // n_embed region of d_out (N*D)

    // degrees -> inv_deg (clamped at 1)
    k_fill0 <<<(N + 255) / 256, 256, 0, stream>>>(deg, N);
    k_deg   <<<(E + 255) / 256, 256, 0, stream>>>(dst, deg, E);
    k_invdeg<<<(N + 255) / 256, 256, 0, stream>>>(deg, N);

    // buffer choreography (z consumed by GEMM1 before GEMM2 writes h_next)
    const float* h = x;
    float* zbuf[3] = {bufA, bufA, nemb};
    float* hout[3] = {nemb, bufA, nemb};

    const int gemmBlocks = (N + 127) / 128;
    for (int l = 0; l < 3; ++l) {
        k_fill0  <<<(int)((ND + 255) / 256), 256, 0, stream>>>(agg, ND);
        k_scatter<<<(E + 7) / 8, 256, 0, stream>>>(h, src, dst, agg, E);
        k_combine<<<(int)((ND / 4 + 255) / 256), 256, 0, stream>>>(
            h, agg, deg, eps, l, zbuf[l], ND / 4);
        k_gemm_lrelu<<<gemmBlocks, 256, 0, stream>>>(
            zbuf[l], W1 + (long)l * DIM * DIM, b1 + (long)l * DIM, agg, N);
        k_gemm_lrelu<<<gemmBlocks, 256, 0, stream>>>(
            agg, W2 + (long)l * DIM * DIM, b2 + (long)l * DIM, hout[l], N);
        h = hout[l];
    }

    k_out<<<(N + 7) / 8, 256, 0, stream>>>(nemb, Wout, bout, out, N);
}
